// InfoNCE_9079560864194
// MI455X (gfx1250) — matrix operand order repaired and checked by reading, not compiled, not run on hardware
//
#include <hip/hip_runtime.h>
#include <math.h>

#define B_ROWS 4096
#define D_DIM  768
#define N2     (2 * B_ROWS)   // 8192 rows in reps
#define INV_T  2.0f           // 1 / temperature(0.5)

typedef __attribute__((ext_vector_type(16))) _Float16 v16h;
typedef __attribute__((ext_vector_type(8)))  _Float16 v8h;
typedef __attribute__((ext_vector_type(8)))  float    v8f;

union FragH { v16h v; v8h h[2]; };

// ---------------------------------------------------------------------------
// Kernel 1: L2-normalize both views into a single f16 reps matrix [8192, 768]
// ---------------------------------------------------------------------------
__global__ __launch_bounds__(256) void nrm_kernel(const float* __restrict__ a,
                                                  const float* __restrict__ b,
                                                  _Float16* __restrict__ reps) {
    const int row = blockIdx.x;                       // 0..8191
    const float* src = (row < B_ROWS) ? (a + (size_t)row * D_DIM)
                                      : (b + (size_t)(row - B_ROWS) * D_DIM);
    const int tid = threadIdx.x;                      // 256 threads, 3 elems each
    float x0 = src[tid], x1 = src[tid + 256], x2 = src[tid + 512];
    float s = x0 * x0 + x1 * x1 + x2 * x2;

    __shared__ float red[8];
    #pragma unroll
    for (int off = 16; off; off >>= 1) s += __shfl_down(s, off, 32);
    if ((tid & 31) == 0) red[tid >> 5] = s;
    __syncthreads();
    if (tid < 8) {
        float t = red[tid];
        #pragma unroll
        for (int off = 4; off; off >>= 1) t += __shfl_down(t, off, 8);
        if (tid == 0) red[0] = t;
    }
    __syncthreads();
    const float inv = 1.0f / fmaxf(sqrtf(red[0]), 1e-12f);
    _Float16* dst = reps + (size_t)row * D_DIM;
    dst[tid]       = (_Float16)(x0 * inv);
    dst[tid + 256] = (_Float16)(x1 * inv);
    dst[tid + 512] = (_Float16)(x2 * inv);
}

// ---------------------------------------------------------------------------
// Kernel 2: fused sim = reps @ reps^T tile GEMM + online exp row-sums.
// One workgroup = 16 rows of sim. 8 waves x 2 column tiles in flight.
// Positive-pair values are gathered branch-free via select-accumulate.
// ---------------------------------------------------------------------------
__global__ __launch_bounds__(256) void sim_kernel(const _Float16* __restrict__ reps,
                                                  float* __restrict__ rowLoss) {
    __shared__ __align__(16) _Float16 aLds[16 * D_DIM];   // 24 KB A tile
    __shared__ float rowSumW[8][16];                      // per-wave partial denoms
    __shared__ float posW[8][16];                         // per-wave positive partials

    const int tid     = threadIdx.x;
    const int wave    = tid >> 5;
    const int lane    = tid & 31;
    const int rowBase = blockIdx.x * 16;

    // Stage A tile (16 rows x 768 f16) into LDS, 16B per transfer.
    {
        const uint4* g = (const uint4*)(reps + (size_t)rowBase * D_DIM);
        uint4* l = (uint4*)aLds;
        #pragma unroll
        for (int i = tid; i < (16 * D_DIM * 2) / 16; i += 256) l[i] = g[i];
    }
    __syncthreads();

    const int mrow  = lane & 15;            // A row / B column within tile
    const int half  = lane >> 4;            // lane half
    const int aOff  = half * 16;            // A K-offset: lane half selects K block of 16 (matches B)
    const int bOff  = half * 16;            // B K-offset: lane half selects K block of 16 (matches A)

    const _Float16* aRow = aLds + mrow * D_DIM;

    float sume[2][8];                       // running exp row-sums
    float posA[8];                          // running positive-pair accumulators
    #pragma unroll
    for (int r = 0; r < 8; ++r) { sume[0][r] = 0.0f; sume[1][r] = 0.0f; posA[r] = 0.0f; }

    // 512 column tiles total; wave w owns pairs (16*it + 2w, +1), it = 0..31
    for (int it = 0; it < 32; ++it) {
        const int ct0  = it * 16 + wave * 2;
        const int col0 = ct0 * 16;
        const int col1 = col0 + 16;
        const _Float16* bRow0 = reps + (size_t)(col0 + mrow) * D_DIM;
        const _Float16* bRow1 = reps + (size_t)(col1 + mrow) * D_DIM;

        if (it + 1 < 32) {  // prefetch next pair's B rows (256 rows ahead)
            __builtin_prefetch(bRow0 + (size_t)256 * D_DIM, 0, 0);
            __builtin_prefetch(bRow1 + (size_t)256 * D_DIM, 0, 0);
        }

        v8f c0 = {}, c1 = {};
        #pragma unroll 4
        for (int k = 0; k < D_DIM; k += 32) {
            FragH fa, fb0, fb1;
            // A 16x32 f16 layout: elements 0-15 = K[k+aOff .. +15] (contiguous, same as B)
            fa.h[0]  = *(const v8h*)(aRow + k + aOff);
            fa.h[1]  = *(const v8h*)(aRow + k + aOff + 8);
            // B 32x16 f16 layout: lane half selects K block of 16 contiguous values
            fb0.h[0] = *(const v8h*)(bRow0 + k + bOff);
            fb0.h[1] = *(const v8h*)(bRow0 + k + bOff + 8);
            fb1.h[0] = *(const v8h*)(bRow1 + k + bOff);
            fb1.h[1] = *(const v8h*)(bRow1 + k + bOff + 8);
            c0 = __builtin_amdgcn_wmma_f32_16x16x32_f16(false, fa.v, false, fb0.v,
                                                        (short)0, c0, false, false);
            c1 = __builtin_amdgcn_wmma_f32_16x16x32_f16(false, fa.v, false, fb1.v,
                                                        (short)0, c1, false, false);
        }

        // C/D layout: VGPR r, lanes 0-15 -> M=r, lanes 16-31 -> M=r+8; N = lane&15
        #pragma unroll
        for (int r = 0; r < 8; ++r) {
            const int grow = rowBase + r + half * 8;
            const int pos  = (grow < B_ROWS) ? grow + B_ROWS : grow - B_ROWS;
            const int gcol0 = col0 + mrow;
            const int gcol1 = col1 + mrow;
            const float v0 = c0[r];
            const float v1 = c1[r];
            // branch-free positive capture: exactly one (lane, tile) matches per row
            posA[r] += (gcol0 == pos) ? v0 : 0.0f;
            posA[r] += (gcol1 == pos) ? v1 : 0.0f;
            // diagonal-masked denominator accumulation
            sume[0][r] += (gcol0 == grow) ? 0.0f : __expf(v0 * INV_T);
            sume[1][r] += (gcol1 == grow) ? 0.0f : __expf(v1 * INV_T);
        }
    }

    // Reduce the 16 lanes sharing each row M, then one slot per wave (no atomics:
    // fixed-order summation keeps the result bitwise deterministic).
    #pragma unroll
    for (int r = 0; r < 8; ++r) {
        float s = sume[0][r] + sume[1][r];
        float p = posA[r];
        #pragma unroll
        for (int off = 8; off; off >>= 1) {
            s += __shfl_xor(s, off, 32);
            p += __shfl_xor(p, off, 32);
        }
        if (mrow == 0) {
            rowSumW[wave][r + half * 8] = s;
            posW[wave][r + half * 8]    = p;
        }
    }
    __syncthreads();

    if (tid < 16) {
        float denom = 0.0f, pos = 0.0f;
        #pragma unroll
        for (int w = 0; w < 8; ++w) { denom += rowSumW[w][tid]; pos += posW[w][tid]; }
        rowLoss[rowBase + tid] = __logf(denom) - pos * INV_T;
    }
}

// ---------------------------------------------------------------------------
// Kernel 3: deterministic tree reduction of the 8192 per-row losses.
// ---------------------------------------------------------------------------
__global__ __launch_bounds__(256) void reduce_kernel(const float* __restrict__ rowLoss,
                                                     float* __restrict__ out) {
    __shared__ float red[256];
    const int tid = threadIdx.x;
    float s = 0.0f;
    for (int i = tid; i < N2; i += 256) s += rowLoss[i];
    red[tid] = s;
    __syncthreads();
    for (int off = 128; off; off >>= 1) {
        if (tid < off) red[tid] += red[tid + off];
        __syncthreads();
    }
    if (tid == 0) out[0] = red[0] / (float)N2;
}

extern "C" void kernel_launch(void* const* d_in, const int* in_sizes, int n_in,
                              void* d_out, int out_size, void* d_ws, size_t ws_size,
                              hipStream_t stream) {
    (void)in_sizes; (void)n_in; (void)out_size; (void)ws_size;
    const float* poly = (const float*)d_in[0];
    const float* cemb = (const float*)d_in[1];

    _Float16* reps    = (_Float16*)d_ws;                                 // 12.58 MB
    float*    rowLoss = (float*)((char*)d_ws +
                                 (size_t)N2 * D_DIM * sizeof(_Float16)); // +32 KB

    nrm_kernel<<<N2, 256, 0, stream>>>(poly, cemb, reps);
    sim_kernel<<<N2 / 16, 256, 0, stream>>>(reps, rowLoss);
    reduce_kernel<<<1, 256, 0, stream>>>(rowLoss, (float*)d_out);
}
